// VisionTransformer3D_36773509988797
// MI455X (gfx1250) — matrix operand order repaired and checked by reading, not compiled, not run on hardware
//
#include <hip/hip_runtime.h>
#include <hip/hip_bf16.h>
#include <math.h>

#define BATCH   2
#define EMB     384
#define NHEAD   8
#define HDIM    48
#define NLAYER  6
#define MLP_DIM 1536
#define NPATCH  1728
#define NTOK    1729
#define MTOK    (BATCH*NTOK)   // 3458
#define KPAD    1760           // keys padded to 55*32
#define HDP     64             // head dim padded for QK k-loop
#define QTILES  109            // ceil(1729/16)
#define KBLKS   55             // ceil(1729/32)

typedef _Float16 v16h __attribute__((ext_vector_type(16)));
typedef _Float16 v8h  __attribute__((ext_vector_type(8)));
typedef float    v8f  __attribute__((ext_vector_type(8)));

union V16H { v16h v; v8h h[2]; };

__device__ __forceinline__ v8f wmma16(v16h a, v16h b, v8f c) {
  // D = A(16x32 f16) * B(32x16 f16) + C(16x16 f32)
  return __builtin_amdgcn_wmma_f32_16x16x32_f16(false, a, false, b, (short)0, c, false, false);
}

__device__ __forceinline__ v16h load16(const _Float16* p) {
  V16H u; u.h[0] = *(const v8h*)p; u.h[1] = *(const v8h*)(p + 8); return u.v;
}

// A-fragment (16x32 f16): lanes 0-15 -> M=lane, K 0-15 ; lanes 16-31 -> K 16-31
__device__ __forceinline__ v16h a_frag(const _Float16* base, int row0, int ld, int lane) {
  int l  = lane & 15;
  int ko = (lane & 16) ? 16 : 0;
  const _Float16* p = base + (size_t)(row0 + l) * ld;
  V16H u; u.h[0] = *(const v8h*)(p + ko); u.h[1] = *(const v8h*)(p + ko + 8);
  return u.v;
}

// B-fragment (32x16 f16): col N = lane&15 ; lanes 0-15 -> K 0-15, lanes 16-31 -> K 16-31
__device__ __forceinline__ v16h b_frag(const _Float16* base, int col0, int ld, int lane) {
  int l  = lane & 15;
  int ko = (lane & 16) ? 16 : 0;
  return load16(base + (size_t)(col0 + l) * ld + ko);
}

// CDNA5 async copy: 16 bytes global -> LDS, tracked with ASYNCcnt (bypasses VGPRs)
__device__ __forceinline__ void async_b128(unsigned lds_off, unsigned long long gaddr) {
  asm volatile("global_load_async_to_lds_b128 %0, %1, off"
               :: "v"(lds_off), "v"(gaddr) : "memory");
}

// ---------------------------------------------------------------------------
// f16 GEMM: C[M,Nout] = A[M,K]f16 * W[Nout,K]f16^T + bias (+res)(gelu)(f16 out)
// 128x64 block tile, 8 waves x (32x32), double-buffered async-LDS staging.
// ---------------------------------------------------------------------------
template<bool GELU, bool RES, bool OUT16>
__global__ __launch_bounds__(256) void gemm_f16_k(
    const _Float16* __restrict__ A, const _Float16* __restrict__ W,
    const float* __restrict__ bias, const float* __restrict__ residual,
    void* __restrict__ Cout, int M, int Nout, int K)
{
  __shared__ __align__(16) _Float16 As[2][128*32];
  __shared__ __align__(16) _Float16 Bs[2][64*32];
  const int tid = threadIdx.x, lane = tid & 31, wave = tid >> 5;
  const int brow = blockIdx.x * 128, bcol = blockIdx.y * 64;
  const int wr = (wave >> 1) * 32, wc = (wave & 1) * 32;
  v8f acc[2][2] = {};
  const int nk = K >> 5;

  auto prefetch = [&](int kt, int buf) {
    const int k0 = kt << 5;
#pragma unroll
    for (int i = 0; i < 2; ++i) {          // A tile: 128 rows x 64B = 512 chunks
      int c = tid + i * 256;
      int r = c >> 2, part = c & 3;
      int gr = brow + r; gr = (gr < M) ? gr : (M - 1);   // clamp: rows>=M discarded later
      unsigned lds = (unsigned)(size_t)(void*)&As[buf][r*32 + part*8];
      async_b128(lds, (unsigned long long)(size_t)(A + (size_t)gr*K + k0 + part*8));
    }
    {                                      // B tile: 64 rows x 64B = 256 chunks
      int r = tid >> 2, part = tid & 3;
      unsigned lds = (unsigned)(size_t)(void*)&Bs[buf][r*32 + part*8];
      async_b128(lds, (unsigned long long)(size_t)(W + (size_t)(bcol + r)*K + k0 + part*8));
    }
  };

  prefetch(0, 0);
  for (int kt = 0; kt < nk; ++kt) {
    const int buf = kt & 1;
    if (kt + 1 < nk) {
      prefetch(kt + 1, buf ^ 1);           // overlap next copy with this tile's WMMAs
      asm volatile("s_wait_asynccnt 3" ::: "memory");   // in-order: first 3 (tile kt) done
    } else {
      asm volatile("s_wait_asynccnt 0" ::: "memory");
    }
    __syncthreads();
#pragma unroll
    for (int i = 0; i < 2; ++i) {
      v16h a = a_frag(&As[buf][0], wr + i*16, 32, lane);
#pragma unroll
      for (int j = 0; j < 2; ++j) {
        v16h b = b_frag(&Bs[buf][0], wc + j*16, 32, lane);
        acc[i][j] = wmma16(a, b, acc[i][j]);
      }
    }
    __syncthreads();
  }
  const int l = lane & 15;
  const int mof = (lane & 16) ? 8 : 0;
#pragma unroll
  for (int i = 0; i < 2; ++i)
#pragma unroll
    for (int j = 0; j < 2; ++j) {
      int col = bcol + wc + j*16 + l;
      if (col < Nout) {
#pragma unroll
        for (int r = 0; r < 8; ++r) {
          int row = brow + wr + i*16 + mof + r;
          if (row < M) {
            float v = acc[i][j][r] + bias[col];
            if (GELU) v = 0.5f * v * (1.f + erff(v * 0.70710678f));
            if (RES)  v += residual[(size_t)row*Nout + col];
            if (OUT16) ((_Float16*)Cout)[(size_t)row*Nout + col] = (_Float16)v;
            else       ((float*)Cout)[(size_t)row*Nout + col] = v;
          }
        }
      }
    }
}

// ---------------------------------------------------------------------------
// Patch embedding GEMM: patches[3456,512](gathered f32) * conv_w[384,512]^T
// fused +conv_b +pos_embed -> t (f32 residual stream)
// ---------------------------------------------------------------------------
__global__ __launch_bounds__(256) void patch_embed_k(
    const float* __restrict__ x, const float* __restrict__ conv_w,
    const float* __restrict__ conv_b, const float* __restrict__ pos,
    float* __restrict__ t)
{
  __shared__ __align__(16) _Float16 As[128*32];
  __shared__ __align__(16) _Float16 Bs[64*32];
  const int tid = threadIdx.x, lane = tid & 31, wave = tid >> 5;
  const int brow = blockIdx.x * 128, bcol = blockIdx.y * 64;
  const int wr = (wave >> 1) * 32, wc = (wave & 1) * 32;
  v8f acc[2][2] = {};

  for (int k0 = 0; k0 < 512; k0 += 32) {
    for (int idx = tid; idx < 128*32; idx += 256) {
      int r = idx >> 5, c = idx & 31, gr = brow + r;   // gr < 3456 (27 blocks)
      int b = gr / NPATCH, pid = gr % NPATCH;
      int kk = k0 + c;
      int dz = kk >> 6, dy = (kk >> 3) & 7, dx = kk & 7;
      int pz = pid / 144, py = (pid / 12) % 12, px = pid % 12;
      size_t off = (size_t)b*96*96*96 +
                   ((size_t)(pz*8+dz)*96 + (py*8+dy))*96 + (px*8+dx);
      As[idx] = (_Float16)x[off];
    }
    for (int idx = tid; idx < 64*32; idx += 256) {
      int r = idx >> 5, c = idx & 31;
      Bs[idx] = (_Float16)conv_w[(size_t)(bcol + r)*512 + k0 + c];
    }
    __syncthreads();
#pragma unroll
    for (int i = 0; i < 2; ++i) {
      v16h a = a_frag(As, wr + i*16, 32, lane);
#pragma unroll
      for (int j = 0; j < 2; ++j) {
        v16h b = b_frag(Bs, wc + j*16, 32, lane);
        acc[i][j] = wmma16(a, b, acc[i][j]);
      }
    }
    __syncthreads();
  }
  const int l = lane & 15;
  const int mof = (lane & 16) ? 8 : 0;
#pragma unroll
  for (int i = 0; i < 2; ++i)
#pragma unroll
    for (int j = 0; j < 2; ++j) {
      int col = bcol + wc + j*16 + l;
#pragma unroll
      for (int r = 0; r < 8; ++r) {
        int row = brow + wr + i*16 + mof + r;
        int b = row / NPATCH, tok = 1 + row % NPATCH;
        float v = acc[i][j][r] + conv_b[col] + pos[(size_t)tok*EMB + col];
        t[((size_t)b*NTOK + tok)*EMB + col] = v;
      }
    }
}

__global__ void cls_pos_k(const float* __restrict__ cls, const float* __restrict__ pos,
                          float* __restrict__ t) {
  int i = blockIdx.x * blockDim.x + threadIdx.x;
  if (i >= BATCH*EMB) return;
  int b = i / EMB, e = i % EMB;
  t[(size_t)b*NTOK*EMB + e] = cls[e] + pos[e];
}

__global__ void f32_to_f16_k(const float* __restrict__ s, _Float16* __restrict__ d, size_t n) {
  size_t i = (size_t)blockIdx.x * blockDim.x + threadIdx.x;
  if (i < n) d[i] = (_Float16)s[i];
}

// ---------------------------------------------------------------------------
// LayerNorm over EMB=384; one wave per row; f32 in, f16 out (GEMM operand)
// ---------------------------------------------------------------------------
__global__ __launch_bounds__(256) void layernorm_k(
    const float* __restrict__ in, const float* __restrict__ w,
    const float* __restrict__ b, _Float16* __restrict__ out, int rows)
{
  int wave = threadIdx.x >> 5, lane = threadIdx.x & 31;
  int row = blockIdx.x * 8 + wave;
  if (row >= rows) return;
  const float* p = in + (size_t)row * EMB;
  float vals[12]; float s = 0.f;
#pragma unroll
  for (int i = 0; i < 12; ++i) { vals[i] = p[lane + i*32]; s += vals[i]; }
#pragma unroll
  for (int off = 16; off; off >>= 1) s += __shfl_xor(s, off);
  float mu = s * (1.f/EMB);
  float v = 0.f;
#pragma unroll
  for (int i = 0; i < 12; ++i) { float d = vals[i] - mu; v += d*d; }
#pragma unroll
  for (int off = 16; off; off >>= 1) v += __shfl_xor(v, off);
  float rs = rsqrtf(v * (1.f/EMB) + 1e-5f);
  _Float16* q = out + (size_t)row * EMB;
#pragma unroll
  for (int i = 0; i < 12; ++i) {
    int e = lane + i*32;
    q[e] = (_Float16)((vals[i] - mu) * rs * w[e] + b[e]);
  }
}

// ---------------------------------------------------------------------------
// qkv[B,N,1152] f32 -> qh/kh [bh][KPAD][64] f16 (q pre-scaled), vT [bh][64][KPAD]
// ---------------------------------------------------------------------------
__global__ void qkv_split_k(const float* __restrict__ qkv,
                            _Float16* __restrict__ qh, _Float16* __restrict__ kh,
                            _Float16* __restrict__ vT)
{
  size_t idx = (size_t)blockIdx.x * blockDim.x + threadIdx.x;
  if (idx >= (size_t)BATCH*NHEAD*KPAD*HDP) return;
  int d  = idx & 63;
  int n  = (int)((idx >> 6) % KPAD);
  int bh = (int)(idx / ((size_t)KPAD*HDP));
  int b = bh >> 3, hh = bh & 7;
  float q = 0.f, k = 0.f, v = 0.f;
  if (n < NTOK && d < HDIM) {
    const float* base = qkv + ((size_t)(b*NTOK + n))*(3*EMB) + hh*HDIM + d;
    q = base[0]   * 0.14433756729740645f;  // HD^-0.5
    k = base[EMB];
    v = base[2*EMB];
  }
  qh[idx] = (_Float16)q;
  kh[idx] = (_Float16)k;
  vT[((size_t)bh*HDP + d)*KPAD + n] = (_Float16)v;
}

// ---------------------------------------------------------------------------
// Flash attention: one wave per (b,h,16-query tile); online softmax over
// 32-key blocks. 4 WMMAs for S (K padded to 64), 3 WMMAs for P*V (48 cols).
// Emits f16 output so the proj GEMM consumes it directly.
// ---------------------------------------------------------------------------
__global__ __launch_bounds__(128) void attn_flash_k(
    const _Float16* __restrict__ qh, const _Float16* __restrict__ kh,
    const _Float16* __restrict__ vT, _Float16* __restrict__ o)
{
  __shared__ __align__(16) _Float16 Pl[4][16*32];
  const int wave = threadIdx.x >> 5, lane = threadIdx.x & 31;
  const int gw = blockIdx.x * 4 + wave;
  const int bh = gw / QTILES, qt = gw % QTILES;
  if (bh >= BATCH*NHEAD) return;
  const int q0 = qt * 16;
  const _Float16* Q  = qh + (size_t)bh * KPAD * HDP;
  const _Float16* Kp = kh + (size_t)bh * KPAD * HDP;
  const _Float16* Vp = vT + (size_t)bh * HDP * KPAD;
  const int l   = lane & 15;
  const int bk0 = (lane & 16) ? 16 : 0;

  v16h qa[2];
#pragma unroll
  for (int t = 0; t < 2; ++t) qa[t] = a_frag(Q + t*32, q0, HDP, lane);

  v8f oacc[3] = {};
  float mrow[8], lrow[8];
#pragma unroll
  for (int r = 0; r < 8; ++r) { mrow[r] = -1e30f; lrow[r] = 0.f; }

  for (int kb = 0; kb < KBLKS; ++kb) {
    const int k0 = kb * 32;
    v8f S[2] = {};
#pragma unroll
    for (int j = 0; j < 2; ++j) {
      const _Float16* pk = Kp + (size_t)(k0 + j*16 + l) * HDP;
#pragma unroll
      for (int t = 0; t < 2; ++t) {
        v16h bf = load16(pk + t*32 + bk0);
        S[j] = wmma16(qa[t], bf, S[j]);
      }
    }
#pragma unroll
    for (int j = 0; j < 2; ++j)
      if (k0 + j*16 + l >= NTOK) {
#pragma unroll
        for (int r = 0; r < 8; ++r) S[j][r] = -1e30f;
      }
    float alpha[8];
#pragma unroll
    for (int r = 0; r < 8; ++r) {
      float mx = fmaxf(S[0][r], S[1][r]);
#pragma unroll
      for (int off = 1; off < 16; off <<= 1) mx = fmaxf(mx, __shfl_xor(mx, off));
      float mnew = fmaxf(mrow[r], mx);
      alpha[r] = __expf(mrow[r] - mnew);
      mrow[r] = mnew;
      float p0 = __expf(S[0][r] - mnew);
      float p1 = __expf(S[1][r] - mnew);
      float ps = p0 + p1;
#pragma unroll
      for (int off = 1; off < 16; off <<= 1) ps += __shfl_xor(ps, off);
      lrow[r] = lrow[r] * alpha[r] + ps;
      S[0][r] = p0; S[1][r] = p1;
    }
#pragma unroll
    for (int t = 0; t < 3; ++t)
#pragma unroll
      for (int r = 0; r < 8; ++r) oacc[t][r] = oacc[t][r] * alpha[r];

    _Float16* Pw = Pl[wave];
    const int mof = (lane & 16) ? 8 : 0;
#pragma unroll
    for (int r = 0; r < 8; ++r) {
      Pw[(mof + r)*32 + l]      = (_Float16)S[0][r];
      Pw[(mof + r)*32 + 16 + l] = (_Float16)S[1][r];
    }
    asm volatile("s_wait_dscnt 0" ::: "memory");
    v16h pa = a_frag(Pw, 0, 32, lane);
#pragma unroll
    for (int t = 0; t < 3; ++t) {
      const _Float16* pv = Vp + (size_t)(t*16 + l) * KPAD + k0 + bk0;
      v16h bf = load16(pv);
      oacc[t] = wmma16(pa, bf, oacc[t]);
    }
  }
  const int mof = (lane & 16) ? 8 : 0;
  const int b = bh >> 3, hh = bh & 7;
#pragma unroll
  for (int t = 0; t < 3; ++t) {
    int col = t*16 + l;
#pragma unroll
    for (int r = 0; r < 8; ++r) {
      int qrow = q0 + mof + r;
      if (qrow < NTOK)
        o[((size_t)(b*NTOK + qrow))*EMB + hh*HDIM + col] = (_Float16)(oacc[t][r] / lrow[r]);
    }
  }
}

// ---------------------------------------------------------------------------
// Head: logits[b,c] = ln(t)[b,0,:] . head_w[c,:] + head_b[c]
// ---------------------------------------------------------------------------
__global__ void head_k(const _Float16* __restrict__ tln, const float* __restrict__ hw,
                       const float* __restrict__ hb, float* __restrict__ out) {
  int wave = threadIdx.x >> 5, lane = threadIdx.x & 31;
  if (wave >= BATCH*2) return;
  int b = wave >> 1, c = wave & 1;
  const _Float16* tr = tln + (size_t)(b * NTOK) * EMB;
  const float* w = hw + (size_t)c * EMB;
  float s = 0.f;
  for (int i = lane; i < EMB; i += 32) s += (float)tr[i] * w[i];
#pragma unroll
  for (int off = 16; off; off >>= 1) s += __shfl_xor(s, off);
  if (lane == 0) out[b*2 + c] = s + hb[c];
}

// ---------------------------------------------------------------------------
extern "C" void kernel_launch(void* const* d_in, const int* in_sizes, int n_in,
                              void* d_out, int out_size, void* d_ws, size_t ws_size,
                              hipStream_t stream) {
  (void)in_sizes; (void)n_in; (void)out_size; (void)ws_size;
  const float* x      = (const float*)d_in[0];
  const float* conv_w = (const float*)d_in[1];
  const float* conv_b = (const float*)d_in[2];
  const float* cls_t  = (const float*)d_in[3];
  const float* pos    = (const float*)d_in[4];
  const float* ln1_w  = (const float*)d_in[5];
  const float* ln1_b  = (const float*)d_in[6];
  const float* qkv_w  = (const float*)d_in[7];
  const float* qkv_b  = (const float*)d_in[8];
  const float* proj_w = (const float*)d_in[9];
  const float* proj_b = (const float*)d_in[10];
  const float* ln2_w  = (const float*)d_in[11];
  const float* ln2_b  = (const float*)d_in[12];
  const float* mlp_w1 = (const float*)d_in[13];
  const float* mlp_b1 = (const float*)d_in[14];
  const float* mlp_w2 = (const float*)d_in[15];
  const float* mlp_b2 = (const float*)d_in[16];
  const float* norm_w = (const float*)d_in[17];
  const float* norm_b = (const float*)d_in[18];
  const float* head_w = (const float*)d_in[19];
  const float* head_b = (const float*)d_in[20];
  float* out = (float*)d_out;

  // workspace layout (every region size is a multiple of 16 bytes)
  float* t    = (float*)d_ws;                         // [MTOK,EMB] residual (f32)
  float* qkvb = t + (size_t)MTOK*EMB;                 // [MTOK,1152] f32
  _Float16* hh  = (_Float16*)(qkvb + (size_t)MTOK*3*EMB); // LN out f16
  _Float16* obh = hh  + (size_t)MTOK*EMB;             // attn out f16
  _Float16* mbh = obh + (size_t)MTOK*EMB;             // gelu out f16
  _Float16* qh  = mbh + (size_t)MTOK*MLP_DIM;
  _Float16* kh  = qh  + (size_t)BATCH*NHEAD*KPAD*HDP;
  _Float16* vT  = kh  + (size_t)BATCH*NHEAD*KPAD*HDP;
  _Float16* wq  = vT  + (size_t)BATCH*NHEAD*KPAD*HDP; // f16 weights
  _Float16* wp  = wq  + (size_t)NLAYER*3*EMB*EMB;
  _Float16* w1  = wp  + (size_t)NLAYER*EMB*EMB;
  _Float16* w2  = w1  + (size_t)NLAYER*MLP_DIM*EMB;

  const int mtiles = (MTOK + 127) / 128;  // 28
  const int lnblks = (MTOK + 7) / 8;      // 433

  // one-time (per call) f32 -> f16 weight conversion
  {
    size_t nq = (size_t)NLAYER*3*EMB*EMB, np = (size_t)NLAYER*EMB*EMB,
           n1 = (size_t)NLAYER*MLP_DIM*EMB, n2 = (size_t)NLAYER*EMB*MLP_DIM;
    f32_to_f16_k<<<(unsigned)((nq+255)/256), 256, 0, stream>>>(qkv_w, wq, nq);
    f32_to_f16_k<<<(unsigned)((np+255)/256), 256, 0, stream>>>(proj_w, wp, np);
    f32_to_f16_k<<<(unsigned)((n1+255)/256), 256, 0, stream>>>(mlp_w1, w1, n1);
    f32_to_f16_k<<<(unsigned)((n2+255)/256), 256, 0, stream>>>(mlp_w2, w2, n2);
  }

  patch_embed_k<<<dim3(NPATCH*BATCH/128, EMB/64), 256, 0, stream>>>(x, conv_w, conv_b, pos, t);
  cls_pos_k<<<(BATCH*EMB + 255)/256, 256, 0, stream>>>(cls_t, pos, t);

  for (int i = 0; i < NLAYER; ++i) {
    layernorm_k<<<lnblks, 256, 0, stream>>>(t, ln1_w + i*EMB, ln1_b + i*EMB, hh, MTOK);
    gemm_f16_k<false,false,false><<<dim3(mtiles, 3*EMB/64), 256, 0, stream>>>(
        hh, wq + (size_t)i*3*EMB*EMB, qkv_b + (size_t)i*3*EMB, nullptr, qkvb,
        MTOK, 3*EMB, EMB);
    {
      size_t tot = (size_t)BATCH*NHEAD*KPAD*HDP;
      qkv_split_k<<<(unsigned)((tot + 255)/256), 256, 0, stream>>>(qkvb, qh, kh, vT);
    }
    attn_flash_k<<<(BATCH*NHEAD*QTILES)/4, 128, 0, stream>>>(qh, kh, vT, obh);
    gemm_f16_k<false,true,false><<<dim3(mtiles, EMB/64), 256, 0, stream>>>(
        obh, wp + (size_t)i*EMB*EMB, proj_b + (size_t)i*EMB, t, t,
        MTOK, EMB, EMB);
    layernorm_k<<<lnblks, 256, 0, stream>>>(t, ln2_w + i*EMB, ln2_b + i*EMB, hh, MTOK);
    gemm_f16_k<true,false,true><<<dim3(mtiles, MLP_DIM/64), 256, 0, stream>>>(
        hh, w1 + (size_t)i*MLP_DIM*EMB, mlp_b1 + (size_t)i*MLP_DIM, nullptr, mbh,
        MTOK, MLP_DIM, EMB);
    gemm_f16_k<false,true,false><<<dim3(mtiles, EMB/64), 256, 0, stream>>>(
        mbh, w2 + (size_t)i*EMB*MLP_DIM, mlp_b2 + (size_t)i*EMB, t, t,
        MTOK, EMB, MLP_DIM);
  }
  layernorm_k<<<lnblks, 256, 0, stream>>>(t, norm_w, norm_b, hh, MTOK);
  head_k<<<1, 128, 0, stream>>>(hh, head_w, head_b, out);
}
